// NeuralSparse_45921790328878
// MI455X (gfx1250) — matrix operand order repaired and checked by reading, not compiled, not run on hardware
//
#include <hip/hip_runtime.h>
#include <stdint.h>

#define DFEAT 128
#define KSEL  5

typedef __attribute__((ext_vector_type(16))) __bf16       v16bf;
typedef __attribute__((ext_vector_type(8)))  float        v8f;
typedef __attribute__((ext_vector_type(4)))  unsigned int u32x4;

union BFTile { v16bf v; u32x4 q[2]; };

__device__ __forceinline__ uint16_t f32_to_bf16(float f) {
  union { float f; uint32_t u; } c; c.f = f;
  uint32_t r = c.u + 0x7FFFu + ((c.u >> 16) & 1u);   // round-to-nearest-even
  return (uint16_t)(r >> 16);
}

// ---------------- 1) x: f32 -> bf16 (makes gather table L2-resident, 25.6MB) --------
__global__ void ns_convert_x(const float* __restrict__ x, uint16_t* __restrict__ xb, int n) {
  int i = blockIdx.x * blockDim.x + threadIdx.x;
  if (i < n) xb[i] = f32_to_bf16(x[i]);
}

// ---------------- 2) W1 (256x128 f32, row-major) -> bf16 in WMMA B-operand layout ---
// Flat layout: [nt][kt][lane][16 elems]; lane<16: N=nt*16+lane, K=kt*32+{0..7,16..23}[j]
//                                        lane>=16: N=nt*16+(lane-16), K=kt*32+{8..15,24..31}[j]
__global__ void ns_prep_w1(const float* __restrict__ W1, uint16_t* __restrict__ w1s) {
  int idx = blockIdx.x * blockDim.x + threadIdx.x;   // 8*8*32*16 = 32768
  if (idx >= 8 * 8 * 32 * 16) return;
  int j    = idx & 15;
  int lane = (idx >> 4) & 31;
  int kt   = (idx >> 9) & 7;
  int nt   = idx >> 12;
  int K = kt * 32 + (lane >> 4) * 8 + (j & 7) + (j >> 3) * 16;
  int N = nt * 16 + (lane & 15);
  w1s[idx] = f32_to_bf16(W1[K * DFEAT + N]);
}

// ---------------- 3) init output + segment state ------------------------------------
__global__ void ns_init(float* __restrict__ weights, unsigned long long* __restrict__ seg64,
                        int E, int KN) {
  int i = blockIdx.x * blockDim.x + threadIdx.x;
  if (i < E)  weights[i] = 0.0f;
  if (i < KN) seg64[i]   = 0ULL;
}

// ---------------- 4) edge MLP via WMMA ----------------------------------------------
// One wave handles 16 edges. A = 16(edges) x 256(feat concat) bf16 gathered from
// global (VMEM pipe); B = pre-swizzled W1 staged once per block into LDS (DS pipe),
// so the irregular gathers get the VMEM pipe to themselves and WMMA B operands come
// from low-latency ds_load_b128 that co-executes with the matrix pipe.
__global__ __launch_bounds__(256) void ns_mlp_scores(
    const uint16_t* __restrict__ xb, const uint16_t* __restrict__ w1s,
    const float* __restrict__ b1, const float* __restrict__ w2, const float* __restrict__ b2,
    const int* __restrict__ intra, const int* __restrict__ cross,
    int Eintra, int E, float* __restrict__ scores)
{
  __shared__ __align__(16) uint16_t w1lds[8 * 8 * 32 * 16];   // 64 KB of 320 KB/WGP

  // cooperative stage: 4096 x b128 across 256 threads
  {
    const u32x4* src = (const u32x4*)w1s;
    u32x4*       dst = (u32x4*)w1lds;
#pragma unroll
    for (int i = 0; i < 16; ++i)
      dst[threadIdx.x + 256 * i] = src[threadIdx.x + 256 * i];
  }
  __syncthreads();   // before any wave-uniform exit: every wave participates

  const int lane = threadIdx.x & 31;
  const int wave = threadIdx.x >> 5;
  const int tile = blockIdx.x * 8 + wave;
  const int base = tile * 16;
  if (base >= E) return;                    // wave-uniform: EXEC stays all-1 for WMMA
  const int row = lane & 15;                // edge row within tile / N within n-tile
  const int hi  = lane >> 4;                // K-half selector for A layout

  int e = base + row;
  int sIdx = 0, dIdx = 0;
  if (e < E) {
    const int* p = (e < Eintra) ? (intra + 2 * (size_t)e)
                                : (cross + 2 * (size_t)(e - Eintra));
    sIdx = p[0]; dIdx = p[1];
  }
  const uint16_t* sRow = xb + (size_t)sIdx * DFEAT;
  const uint16_t* dRow = xb + (size_t)dIdx * DFEAT;

  // Gather A once (16x256 bf16 across the wave), keep in registers for all 8 n-tiles.
  // A layout (16-bit, 16x32): lane<16 holds row, K {0..7,16..23}; lane>=16: {8..15,24..31}
  BFTile A[8];
#pragma unroll
  for (int kt = 0; kt < 8; ++kt) {
    const uint16_t* rp = (kt < 4) ? sRow : dRow;       // feat 0..127 = src, 128..255 = dst
    const int f0 = (kt & 3) * 32 + hi * 8;
    A[kt].q[0] = *(const u32x4*)(rp + f0);
    A[kt].q[1] = *(const u32x4*)(rp + f0 + 16);
  }

  float s[8];
#pragma unroll
  for (int r = 0; r < 8; ++r) s[r] = 0.0f;

#pragma unroll
  for (int nt = 0; nt < 8; ++nt) {
    v8f acc = {0.f, 0.f, 0.f, 0.f, 0.f, 0.f, 0.f, 0.f};
    const uint16_t* wp = w1lds + ((nt * 8) * 32 + lane) * 16;
#pragma unroll
    for (int kt = 0; kt < 8; ++kt) {
      BFTile B;
      B.q[0] = *(const u32x4*)(wp);
      B.q[1] = *(const u32x4*)(wp + 8);
      wp += 32 * 16;
      acc = __builtin_amdgcn_wmma_f32_16x16x32_bf16(false, A[kt].v, false, B.v,
                                                    (short)0, acc, false, false);
    }
    // C layout: VGPR r, lane<16 -> (M=r, N=lane); lane>=16 -> (M=r+8, N=lane-16)
    const int   n   = nt * 16 + row;
    const float b1v = b1[n];
    const float w2v = w2[n];
#pragma unroll
    for (int r = 0; r < 8; ++r) {
      float h = acc[r] + b1v;
      h = h > 0.f ? h : 0.f;
      s[r] += h * w2v;
    }
  }
  // reduce dot-product partials across the 16 N-lanes of each half-wave
#pragma unroll
  for (int m = 1; m < 16; m <<= 1) {
#pragma unroll
    for (int r = 0; r < 8; ++r) s[r] += __shfl_xor(s[r], m, 32);
  }
  if (row == 0) {                 // lane 0 -> edges base..base+7, lane 16 -> base+8..15
    const float b2v = b2[0];
#pragma unroll
    for (int r = 0; r < 8; ++r) {
      int eo = base + hi * 8 + r;
      if (eo < E) scores[eo] = s[r] + b2v;
    }
  }
}

// ---------------- 5) per-(k,src) argmax via packed u64 atomicMax --------------------
// key = order-preserving uint of logit; low32 = 0xFFFFFFFF - e  => ties pick min e.
__global__ void ns_select(const float* __restrict__ scores,
                          const int* __restrict__ intra, const int* __restrict__ cross,
                          int Eintra, int E, int Nn,
                          unsigned long long* __restrict__ seg64)
{
  int e = blockIdx.x * blockDim.x + threadIdx.x;
  int k = blockIdx.y;
  if (e >= E) return;
  const int* p = (e < Eintra) ? (intra + 2 * (size_t)e)
                              : (cross + 2 * (size_t)(e - Eintra));
  int   src = p[0];
  float sc  = scores[e];
  // deterministic splitmix64 -> Gumbel(0,1)
  unsigned long long z = (unsigned long long)k * (unsigned long long)E + (unsigned long long)e;
  z += 0x9E3779B97F4A7C15ULL;
  z ^= z >> 30; z *= 0xBF58476D1CE4E5B9ULL;
  z ^= z >> 27; z *= 0x94D049BB133111EBULL;
  z ^= z >> 31;
  float u = ((float)(z >> 40) + 0.5f) * (1.0f / 16777216.0f);   // (0,1)
  float g = -__logf(-__logf(u));
  float logit = sc + g;                                          // TAU = 1
  union { float f; uint32_t u; } cv; cv.f = logit;
  uint32_t key = (cv.u & 0x80000000u) ? ~cv.u : (cv.u | 0x80000000u);
  unsigned long long val =
      ((unsigned long long)key << 32) | (unsigned long long)(0xFFFFFFFFu - (uint32_t)e);
  atomicMax(&seg64[(size_t)k * (size_t)Nn + (size_t)src], val);
}

// ---------------- 6) winner scatter: weights[e] += 1 per (k, node) -------------------
__global__ void ns_count(const unsigned long long* __restrict__ seg64, int KN, int E,
                         float* __restrict__ weights)
{
  int i = blockIdx.x * blockDim.x + threadIdx.x;
  if (i >= KN) return;
  unsigned long long v = seg64[i];
  if (v != 0ULL) {
    uint32_t e = 0xFFFFFFFFu - (uint32_t)(v & 0xFFFFFFFFULL);
    if (e < (uint32_t)E) atomicAdd(&weights[e], 1.0f);
  }
}

// ------------------------------------------------------------------------------------
extern "C" void kernel_launch(void* const* d_in, const int* in_sizes, int n_in,
                              void* d_out, int out_size, void* d_ws, size_t ws_size,
                              hipStream_t stream)
{
  (void)n_in; (void)out_size; (void)ws_size;
  const float* x    = (const float*)d_in[0];
  const float* W1   = (const float*)d_in[1];
  const float* b1   = (const float*)d_in[2];
  const float* W2   = (const float*)d_in[3];
  const float* b2   = (const float*)d_in[4];
  const int* intra  = (const int*)d_in[5];
  const int* cross  = (const int*)d_in[6];
  // d_in[7] node_layers: unused by the reference math

  const int nX     = in_sizes[0];
  const int Nn     = nX / DFEAT;
  const int Eintra = in_sizes[5] / 2;
  const int Ecross = in_sizes[6] / 2;
  const int E      = Eintra + Ecross;
  const int KN     = KSEL * Nn;

  char* ws = (char*)d_ws;
  size_t off = 0;
  uint16_t* xb  = (uint16_t*)(ws + off); off += (((size_t)nX * 2) + 255) & ~(size_t)255;
  uint16_t* w1s = (uint16_t*)(ws + off); off += ((size_t)(8 * 8 * 32 * 16 * 2) + 255) & ~(size_t)255;
  float* scores = (float*)(ws + off);    off += (((size_t)E * 4) + 255) & ~(size_t)255;
  unsigned long long* seg64 = (unsigned long long*)(ws + off);

  float* weights = (float*)d_out;

  ns_convert_x<<<(nX + 255) / 256, 256, 0, stream>>>(x, xb, nX);
  ns_prep_w1<<<(8 * 8 * 32 * 16 + 255) / 256, 256, 0, stream>>>(W1, w1s);

  int initN = (E > KN) ? E : KN;
  ns_init<<<(initN + 255) / 256, 256, 0, stream>>>(weights, seg64, E, KN);

  int tiles = (E + 15) / 16;
  ns_mlp_scores<<<(tiles + 7) / 8, 256, 0, stream>>>(xb, w1s, b1, W2, b2,
                                                     intra, cross, Eintra, E, scores);

  dim3 selg((E + 255) / 256, KSEL);
  ns_select<<<selg, 256, 0, stream>>>(scores, intra, cross, Eintra, E, Nn, seg64);

  ns_count<<<(KN + 255) / 256, 256, 0, stream>>>(seg64, KN, E, weights);
}
